// SpikeFCNet_43344809951327
// MI455X (gfx1250) — matrix operand order repaired and checked by reading, not compiled, not run on hardware
//
#include <hip/hip_runtime.h>
#include <cstdint>
#include <cstddef>

// ---- CDNA5 WMMA types ----
typedef __attribute__((ext_vector_type(16))) __bf16 v16bf;
typedef __attribute__((ext_vector_type(8)))  __bf16 v8bf;
typedef __attribute__((ext_vector_type(8)))  float  v8f;
typedef int v4i_vs __attribute__((vector_size(16)));   // matches async-LDS builtin param

#define DECAY 0.95122942450071400f
#define THETA 1.0f
#define ALPHA 0.01f

#define Bsz 256
#define INs 700
#define INP 704     // IN padded to multiple of 32
#define Hs  1024
#define Os  20
#define OPD 32      // O padded to multiple of 16
#define Ts  100

#define LDW1 712    // LDS row stride (elems) for K=704 strips (+8 pad, bank spread)
#define LDW2 1032   // LDS row stride (elems) for K=1024 strips (+8 pad)

#ifndef __has_builtin
#define __has_builtin(x) 0
#endif
#if __has_builtin(__builtin_amdgcn_global_load_async_to_lds_b128) && \
    __has_builtin(__builtin_amdgcn_s_wait_asynccnt)
#define HAVE_ASYNC_LDS 1
#else
#define HAVE_ASYNC_LDS 0
#endif

#define AS1 __attribute__((address_space(1)))
#define AS3 __attribute__((address_space(3)))

// ---------------------------------------------------------------------------
// WMMA operand loaders (CDNA5 ISA §7.12.2, wave32)
// A 16x32 bf16: elems[0..15]=K(k0+2*hk..), hk=8*(lane>>4)
// B 32x16 bf16 (rows of W[N,K]): elems[0..15]=K(k0+hb..), hb=16*(lane>>4)
// C/D 16x16 f32: vgpr v -> row = v + 8*(lane>>4); col = lane&15
// ---------------------------------------------------------------------------
__device__ __forceinline__ v16bf ldA(const __bf16* __restrict__ base, int ld,
                                     int row, int k0, int hk) {
  const __bf16* p = base + (size_t)row * (size_t)ld + (size_t)(k0 + 2 * hk);
  union { v16bf v; v8bf h[2]; } u;
  u.h[0] = *reinterpret_cast<const v8bf*>(p);
  u.h[1] = *reinterpret_cast<const v8bf*>(p + 8);
  return u.v;
}

__device__ __forceinline__ v16bf ldBg(const __bf16* __restrict__ base, int ld,
                                      int row, int k0, int hb) {
  const __bf16* p = base + (size_t)row * (size_t)ld + (size_t)(k0 + hb);
  union { v16bf v; v8bf h[2]; } u;
  u.h[0] = *reinterpret_cast<const v8bf*>(p);
  u.h[1] = *reinterpret_cast<const v8bf*>(p + 8);
  return u.v;
}

// B operand read from LDS-staged strip (strip-local row, padded stride)
__device__ __forceinline__ v16bf ldBl(const __bf16* lds, int lld,
                                      int row, int k0, int hb) {
  const __bf16* p = lds + row * lld + k0 + hb;
  union { v16bf v; v8bf h[2]; } u;
  u.h[0] = *reinterpret_cast<const v8bf*>(p);
  u.h[1] = *reinterpret_cast<const v8bf*>(p + 8);
  return u.v;
}

__device__ __forceinline__ v8f wmma_bf16(v16bf a, v16bf b, v8f c) {
  return __builtin_amdgcn_wmma_f32_16x16x32_bf16(false, a, false, b,
                                                 (short)0, c, false, false);
}

__device__ __forceinline__ v8f vzero8() {
  v8f z = {0.f, 0.f, 0.f, 0.f, 0.f, 0.f, 0.f, 0.f};
  return z;
}

// ---------------------------------------------------------------------------
// Pipelined 16x64 GEMM strip: acc[0..3] += A(16xK) x B(Kx64)
// A streamed from global (depth-1 reg double buffer), B read from LDS strip
// (depth-1 reg double buffer) so WMMAs only need partial dscnt/loadcnt waits.
// ---------------------------------------------------------------------------
template <int KELEMS, int LLD>
__device__ __forceinline__ void gemm_strip(const __bf16* __restrict__ arow, int ald,
                                           const __bf16* bsm,
                                           int m, int hk, int hb, v8f acc[4]) {
  v16bf aC  = ldA(arow, ald, 0, 0, hk);
  v16bf bC0 = ldBl(bsm, LLD,  0 + m, 0, hb);
  v16bf bC1 = ldBl(bsm, LLD, 16 + m, 0, hb);
  v16bf bC2 = ldBl(bsm, LLD, 32 + m, 0, hb);
  v16bf bC3 = ldBl(bsm, LLD, 48 + m, 0, hb);
  for (int kk = 0; kk < KELEMS / 32 - 1; ++kk) {
    const int k = kk * 32 + 32;
    // issue next-iteration loads before current WMMAs (distinct registers)
    v16bf aN  = ldA(arow, ald, 0, k, hk);
    v16bf bN0 = ldBl(bsm, LLD,  0 + m, k, hb);
    v16bf bN1 = ldBl(bsm, LLD, 16 + m, k, hb);
    v16bf bN2 = ldBl(bsm, LLD, 32 + m, k, hb);
    v16bf bN3 = ldBl(bsm, LLD, 48 + m, k, hb);
    acc[0] = wmma_bf16(aC, bC0, acc[0]);
    acc[1] = wmma_bf16(aC, bC1, acc[1]);
    acc[2] = wmma_bf16(aC, bC2, acc[2]);
    acc[3] = wmma_bf16(aC, bC3, acc[3]);
    aC = aN; bC0 = bN0; bC1 = bN1; bC2 = bN2; bC3 = bN3;
  }
  acc[0] = wmma_bf16(aC, bC0, acc[0]);
  acc[1] = wmma_bf16(aC, bC1, acc[1]);
  acc[2] = wmma_bf16(aC, bC2, acc[2]);
  acc[3] = wmma_bf16(aC, bC3, acc[3]);
}

// Cooperative stage of a [rows x kElems] bf16 strip (row stride gld) into LDS
// with padded row stride lld. 16B chunks round-robin across the block.
__device__ __forceinline__ void stage_strip(const __bf16* __restrict__ gbase, int gld,
                                            __bf16* lds, int lld,
                                            int rows, int kElems,
                                            int tid, int nthreads) {
  const int cpr = kElems >> 3;          // 16B chunks per row
  const int chunks = rows * cpr;
  for (int c = tid; c < chunks; c += nthreads) {
    int r  = c / cpr;
    int kc = (c - r * cpr) << 3;
    const __bf16* g = gbase + (size_t)r * gld + kc;
    __bf16* l = lds + r * lld + kc;
#if HAVE_ASYNC_LDS
    __builtin_amdgcn_global_load_async_to_lds_b128(
        (AS1 v4i_vs*)g, (AS3 v4i_vs*)l, 0, 0);
#else
    *reinterpret_cast<v8bf*>(l) = *reinterpret_cast<const v8bf*>(g);
#endif
  }
}

__device__ __forceinline__ void stage_fence() {
#if HAVE_ASYNC_LDS
  __builtin_amdgcn_s_wait_asynccnt(0);
#endif
  __syncthreads();
}

// ---------------------------------------------------------------------------
// Setup kernels
// ---------------------------------------------------------------------------
__global__ void zero16B(uint4* p, int n) {
  int i = blockIdx.x * blockDim.x + threadIdx.x;
  if (i < n) p[i] = make_uint4(0u, 0u, 0u, 0u);
}

__global__ void cvt_pad(const float* __restrict__ src, __bf16* __restrict__ dst,
                        int srows, int scols, int dcols, int n) {
  int i = blockIdx.x * blockDim.x + threadIdx.x;
  if (i >= n) return;
  int c = i % dcols;
  int r = i / dcols;
  float v = (r < srows && c < scols) ? src[(size_t)r * scols + c] : 0.0f;
  dst[i] = (__bf16)v;
}

// x[B,IN,T] f32 -> xt[T,B,INP] bf16 (K zero-padded)
__global__ void xpose(const float* __restrict__ x, __bf16* __restrict__ xt, int n) {
  int i = blockIdx.x * blockDim.x + threadIdx.x;
  if (i >= n) return;
  int k = i % INP;
  int b = (i / INP) % Bsz;
  int t = i / (INP * Bsz);
  float v = (k < INs) ? x[((size_t)b * INs + k) * Ts + t] : 0.0f;
  xt[i] = (__bf16)v;
}

// ---------------------------------------------------------------------------
// FF1 precompute: ff[(t*B+b), h] = xt[(t*B+b), :] . Wi1[h, :]
// grid: 3200 blocks = 200 row-groups x 16 col-strips; block stages its strip.
// ---------------------------------------------------------------------------
__global__ void __launch_bounds__(256) ff1_gemm(const __bf16* __restrict__ xt,
                                                const __bf16* __restrict__ w,
                                                __bf16* __restrict__ ff) {
  extern __shared__ __align__(16) __bf16 smem[];
  const int tid  = threadIdx.x;
  const int lane = tid & 31;
  const int wave = tid >> 5;
  const int nt   = blockIdx.x & 15;
  const int rtg  = blockIdx.x >> 4;
  const int row0 = (rtg * 8 + wave) * 16;
  const int col0 = nt * 64;
  const int m  = lane & 15;
  const int hk = (lane >> 4) << 3;
  const int hb = (lane >> 4) << 4;

  stage_strip(w + (size_t)col0 * INP, INP, smem, LDW1, 64, INP, tid, 256);
  stage_fence();

  v8f acc[4] = {vzero8(), vzero8(), vzero8(), vzero8()};
  gemm_strip<INP, LDW1>(xt + (size_t)(row0 + m) * INP, INP, smem, m, hk, hb, acc);

  const int rb = row0 + ((lane >> 4) << 3);
#pragma unroll
  for (int i = 0; i < 4; ++i) {
    int c = col0 + i * 16 + m;
#pragma unroll
    for (int v = 0; v < 8; ++v)
      ff[(size_t)(rb + v) * Hs + c] = (__bf16)acc[i][v];
  }
}

// ---------------------------------------------------------------------------
// Layer 1 step: m1 = DECAY*m1*(1-s1p) + ff1[t] + s1p @ Wr1^T ; s1n = spike
// grid 32 x 256; block = 8 waves sharing one 64-col strip of Wr1 (LDS-staged)
// ---------------------------------------------------------------------------
__global__ void __launch_bounds__(256) lif1_step(const __bf16* __restrict__ s1p,
                                                 const __bf16* __restrict__ wr1,
                                                 const __bf16* __restrict__ ff,
                                                 float* __restrict__ m1,
                                                 __bf16* __restrict__ s1n) {
  extern __shared__ __align__(16) __bf16 smem[];
  const int tid  = threadIdx.x;
  const int lane = tid & 31;
  const int wave = tid >> 5;
  const int ns   = blockIdx.x >> 1;
  const int mt   = ((blockIdx.x & 1) << 3) + wave;
  const int row0 = mt * 16, col0 = ns * 64;
  const int m  = lane & 15;
  const int hk = (lane >> 4) << 3;
  const int hb = (lane >> 4) << 4;

  stage_strip(wr1 + (size_t)col0 * Hs, Hs, smem, LDW2, 64, Hs, tid, 256);
  stage_fence();

  v8f acc[4] = {vzero8(), vzero8(), vzero8(), vzero8()};
  gemm_strip<Hs, LDW2>(s1p + (size_t)(row0 + m) * Hs, Hs, smem, m, hk, hb, acc);

  const int rb = row0 + ((lane >> 4) << 3);
#pragma unroll
  for (int i = 0; i < 4; ++i) {
    int c = col0 + i * 16 + m;
#pragma unroll
    for (int v = 0; v < 8; ++v) {
      size_t idx = (size_t)(rb + v) * Hs + c;
      float sp = (float)s1p[idx];
      float mn = DECAY * m1[idx] * (1.0f - sp) + (float)ff[idx] + acc[i][v];
      m1[idx] = mn;
      s1n[idx] = (__bf16)((mn >= THETA) ? 1.0f : 0.0f);
    }
  }
}

// ---------------------------------------------------------------------------
// Layer 2 step: m2 = DECAY*m2*(1-s2p) + s1c @ Wi2^T + s2p @ Wr2^T ; s2n = spike
// Both weight strips staged in LDS (rows 0-63: Wi2, rows 64-127: Wr2).
// Two pipelined K passes accumulate into the same accumulators.
// ---------------------------------------------------------------------------
__global__ void __launch_bounds__(256) lif2_step(const __bf16* __restrict__ s1c,
                                                 const __bf16* __restrict__ s2p,
                                                 const __bf16* __restrict__ wi2,
                                                 const __bf16* __restrict__ wr2,
                                                 float* __restrict__ m2,
                                                 __bf16* __restrict__ s2n) {
  extern __shared__ __align__(16) __bf16 smem[];
  const int tid  = threadIdx.x;
  const int lane = tid & 31;
  const int wave = tid >> 5;
  const int ns   = blockIdx.x >> 1;
  const int mt   = ((blockIdx.x & 1) << 3) + wave;
  const int row0 = mt * 16, col0 = ns * 64;
  const int m  = lane & 15;
  const int hk = (lane >> 4) << 3;
  const int hb = (lane >> 4) << 4;

  stage_strip(wi2 + (size_t)col0 * Hs, Hs, smem, LDW2, 64, Hs, tid, 256);
  stage_strip(wr2 + (size_t)col0 * Hs, Hs, smem + 64 * LDW2, LDW2, 64, Hs, tid, 256);
  stage_fence();

  v8f acc[4] = {vzero8(), vzero8(), vzero8(), vzero8()};
  gemm_strip<Hs, LDW2>(s1c + (size_t)(row0 + m) * Hs, Hs, smem, m, hk, hb, acc);
  gemm_strip<Hs, LDW2>(s2p + (size_t)(row0 + m) * Hs, Hs, smem + 64 * LDW2,
                       m, hk, hb, acc);

  const int rb = row0 + ((lane >> 4) << 3);
#pragma unroll
  for (int i = 0; i < 4; ++i) {
    int c = col0 + i * 16 + m;
#pragma unroll
    for (int v = 0; v < 8; ++v) {
      size_t idx = (size_t)(rb + v) * Hs + c;
      float sp = (float)s2p[idx];
      float mn = DECAY * m2[idx] * (1.0f - sp) + acc[i][v];
      m2[idx] = mn;
      s2n[idx] = (__bf16)((mn >= THETA) ? 1.0f : 0.0f);
    }
  }
}

// ---------------------------------------------------------------------------
// Layer 3 step: m3 = DECAY*m3*(1-s3p) + s2c @ Wo^T ; record s3, m3 (tiny GEMM)
// ---------------------------------------------------------------------------
__global__ void __launch_bounds__(256) lif3_step(const __bf16* __restrict__ s2c,
                                                 const __bf16* __restrict__ wo,
                                                 float* __restrict__ m3,
                                                 float* __restrict__ s3,
                                                 float* __restrict__ outS,
                                                 float* __restrict__ outM) {
  const int lane = threadIdx.x & 31;
  const int wid  = blockIdx.x * (blockDim.x >> 5) + (threadIdx.x >> 5);
  const int mt = wid & 15, nt = wid >> 4;
  const int row0 = mt * 16, col0 = nt * 16;
  const int m  = lane & 15;
  const int hk = (lane >> 4) << 3;
  const int hb = (lane >> 4) << 4;
  const __bf16* arow = s2c + (size_t)(row0 + m) * Hs;
  v8f acc = vzero8();
  v16bf aC = ldA(arow, Hs, 0, 0, hk);
  v16bf bC = ldBg(wo, Hs, col0 + m, 0, hb);
  for (int kk = 0; kk < (Hs / 32) - 1; ++kk) {
    const int k = kk * 32 + 32;
    v16bf aN = ldA(arow, Hs, 0, k, hk);
    v16bf bN = ldBg(wo, Hs, col0 + m, k, hb);
    acc = wmma_bf16(aC, bC, acc);
    aC = aN; bC = bN;
  }
  acc = wmma_bf16(aC, bC, acc);

  const int rb = row0 + ((lane >> 4) << 3);
  const int c  = col0 + m;
#pragma unroll
  for (int v = 0; v < 8; ++v) {
    int r = rb + v;
    size_t idx = (size_t)r * OPD + c;
    float sp = s3[idx];
    float mn = DECAY * m3[idx] * (1.0f - sp) + acc[v];
    float sn = (mn >= THETA) ? 1.0f : 0.0f;
    m3[idx] = mn;
    s3[idx] = sn;
    if (c < Os) {
      outS[(size_t)r * Os + c] = sn;
      outM[(size_t)r * Os + c] = mn;
    }
  }
}

// ---------------------------------------------------------------------------
// spike2time + firing rate. potentials[t] = (t==0 ? 0 : m[t-1])
// ---------------------------------------------------------------------------
__global__ void finalize(const float* __restrict__ outS,
                         const float* __restrict__ outM,
                         float* __restrict__ out) {
  int i = blockIdx.x * blockDim.x + threadIdx.x;
  if (i >= Bsz * Os) return;
  float prod = 1.f, tf = 0.f, spiked = 0.f, rate = 0.f, mx = 0.f;
  for (int t = 0; t < Ts; ++t) {
    float s = outS[(size_t)t * (Bsz * Os) + i];
    float first = s * prod;
    tf += first * (float)t;
    spiked += first;
    prod *= (1.f - s);
    rate += s;
    if (t < Ts - 1) mx = fmaxf(mx, outM[(size_t)t * (Bsz * Os) + i]);
  }
  float nospike = (float)Ts + ALPHA * fmaxf(0.f, THETA - mx);
  out[i] = spiked * tf + (1.f - spiked) * nospike;
  out[Bsz * Os + i] = rate * (1.0f / (float)Ts);
}

// ---------------------------------------------------------------------------
extern "C" void kernel_launch(void* const* d_in, const int* in_sizes, int n_in,
                              void* d_out, int out_size, void* d_ws, size_t ws_size,
                              hipStream_t stream) {
  (void)in_sizes; (void)n_in; (void)out_size; (void)ws_size;
  const float* x   = (const float*)d_in[0];
  const float* Wi1 = (const float*)d_in[1];
  const float* Wr1 = (const float*)d_in[2];
  const float* Wi2 = (const float*)d_in[3];
  const float* Wr2 = (const float*)d_in[4];
  const float* Wo  = (const float*)d_in[5];

  char* ws = (char*)d_ws;
  // zero-initialized state region (contiguous): m1,m2,m3,s3,s1a,s2a
  float*  m1   = (float*) (ws + 0);          // 1,048,576
  float*  m2   = (float*) (ws + 1048576);    // 1,048,576
  float*  m3   = (float*) (ws + 2097152);    // 32,768
  float*  s3   = (float*) (ws + 2129920);    // 32,768
  __bf16* s1a  = (__bf16*)(ws + 2162688);    // 524,288
  __bf16* s2a  = (__bf16*)(ws + 2686976);    // 524,288
  // not zero-initialized
  __bf16* s1b  = (__bf16*)(ws + 3211264);    // 524,288
  __bf16* s2b  = (__bf16*)(ws + 3735552);    // 524,288
  __bf16* wi1b = (__bf16*)(ws + 4259840);    // 1,441,792  [1024,704]
  __bf16* wr1b = (__bf16*)(ws + 5701632);    // 2,097,152  [1024,1024]
  __bf16* wi2b = (__bf16*)(ws + 7798784);    // 2,097,152
  __bf16* wr2b = (__bf16*)(ws + 9895936);    // 2,097,152
  __bf16* wob  = (__bf16*)(ws + 11993088);   // 65,536     [32,1024]
  __bf16* xt   = (__bf16*)(ws + 12058624);   // 36,044,800 [T,B,704]
  __bf16* ff1  = (__bf16*)(ws + 48103424);   // 52,428,800 [T,B,1024]
  float*  outS = (float*) (ws + 100532224);  // 2,048,000  [T,B,20]
  float*  outM = (float*) (ws + 102580224);  // 2,048,000

  zero16B<<<784, 256, 0, stream>>>((uint4*)ws, 200704);

  cvt_pad<<<(Hs * INP) / 256, 256, 0, stream>>>(Wi1, wi1b, Hs, INs, INP, Hs * INP);
  cvt_pad<<<(Hs * Hs) / 256, 256, 0, stream>>>(Wr1, wr1b, Hs, Hs, Hs, Hs * Hs);
  cvt_pad<<<(Hs * Hs) / 256, 256, 0, stream>>>(Wi2, wi2b, Hs, Hs, Hs, Hs * Hs);
  cvt_pad<<<(Hs * Hs) / 256, 256, 0, stream>>>(Wr2, wr2b, Hs, Hs, Hs, Hs * Hs);
  cvt_pad<<<(OPD * Hs) / 256, 256, 0, stream>>>(Wo, wob, Os, Hs, Hs, OPD * Hs);

  xpose<<<(Ts * Bsz * INP) / 256, 256, 0, stream>>>(x, xt, Ts * Bsz * INP);

  // FF1 precompute: dynamic LDS = 64*712*2 bytes
  ff1_gemm<<<3200, 256, 64 * LDW1 * 2, stream>>>(xt, wi1b, ff1);

  for (int t = 0; t < Ts; ++t) {
    const __bf16* s1p = (t & 1) ? s1b : s1a;
    __bf16*       s1n = (t & 1) ? s1a : s1b;
    const __bf16* s2p = (t & 1) ? s2b : s2a;
    __bf16*       s2n = (t & 1) ? s2a : s2b;
    lif1_step<<<32, 256, 64 * LDW2 * 2, stream>>>(
        s1p, wr1b, ff1 + (size_t)t * Bsz * Hs, m1, s1n);
    lif2_step<<<32, 256, 128 * LDW2 * 2, stream>>>(
        s1n, s2p, wi2b, wr2b, m2, s2n);
    lif3_step<<<4, 256, 0, stream>>>(s2n, wob, m3, s3,
                                     outS + (size_t)t * Bsz * Os,
                                     outM + (size_t)t * Bsz * Os);
  }

  finalize<<<(Bsz * Os + 255) / 256, 256, 0, stream>>>(outS, outM, (float*)d_out);
}
